// VectorQuantizerEMA_21371757265042
// MI455X (gfx1250) — matrix operand order repaired and checked by reading, not compiled, not run on hardware
//
#include <hip/hip_runtime.h>
#include <math.h>

// ---------------------------------------------------------------------------
// VQ-VAE EMA quantizer for MI455X (gfx1250, wave32, WMMA + async-LDS copy).
//   N = 65536 rows (z), D = 64, K = 1024 codes.
//   Hot loop: argmin_k ( e_sq[k] - 2 * z.e[k] )  via v_wmma_f32_16x16x32_f16,
//   two independent code-tiles in flight to cover the WMMA->VALU hazard.
// Roofline: GEMM = 8.6 GFLOP; traffic ~50 MB -> ~2.2us @ 23.3 TB/s. f16 WMMA
// pushes the GEMM under the memory bound. Codebook is pre-converted to f16
// once, then staged per-block via GLOBAL_LOAD_ASYNC_TO_LDS_B128 (ASYNCcnt),
// overlapping the z-tile transpose/convert done in VALU.
// ---------------------------------------------------------------------------

typedef _Float16 f16;
typedef __attribute__((ext_vector_type(16))) _Float16 v16h;
typedef __attribute__((ext_vector_type(8)))  _Float16 v8h;
typedef __attribute__((ext_vector_type(8)))  float    v8f;

#define DQ      64            // embedding dim
#define KQ      1024          // codebook size
#define HWQ     4096          // H*W of z_e (64*64)
#define DECAYQ  0.99f
#define EPSQ    1e-5f
#define BETAQ   0.25f

// ---------------------------------------------------------------------------
// Kernel 0: zero workspace accumulators (counts[K], embed_sum[K*D], loss[1]).
// ---------------------------------------------------------------------------
__global__ void k_init(float* ws, int n) {
  int i = blockIdx.x * 256 + threadIdx.x;
  if (i < n) ws[i] = 0.0f;
}

// ---------------------------------------------------------------------------
// Kernel 0b: one-time codebook fp32 -> f16 + e_sq precompute (into d_ws).
// ---------------------------------------------------------------------------
__global__ void k_embf16(const float* __restrict__ emb,
                         f16* __restrict__ embF16,
                         float* __restrict__ esqg)
{
  int i = blockIdx.x * 256 + threadIdx.x;   // K*D threads
  embF16[i] = (f16)emb[i];
  if (i < KQ) {
    float s = 0.0f;
    #pragma unroll 8
    for (int d = 0; d < DQ; ++d) { float e = emb[i * DQ + d]; s += e * e; }
    esqg[i] = s;
  }
}

// ---------------------------------------------------------------------------
// Kernel 1: WMMA distance + argmin.
//  block = 256 threads (8 waves), each wave owns 16 z-rows; block owns 128.
//  LDS: codebook f16 [K][D] (128KB, async-copied) + e_sq f32 [K] (4KB,
//  async-copied) + z tile f16 [128][D] (16KB, VALU-converted transpose).
//  z_e layout is [B, D, H, W]; row n -> z_e[(n>>12)*D*HW + d*HW + (n&4095)].
// ---------------------------------------------------------------------------
__global__ void k_argmin(const float* __restrict__ zg,
                         const f16*   __restrict__ embF16,
                         const float* __restrict__ esqg,
                         float* __restrict__ outIdxF,
                         int*   __restrict__ idxWs,
                         float* __restrict__ counts)
{
  extern __shared__ char smem[];
  f16*   le  = (f16*)smem;                          // [K][D] codebook, f16
  float* esq = (float*)(smem + KQ * DQ * 2);        // [K] squared norms
  f16*   lz  = (f16*)(smem + KQ * DQ * 2 + KQ * 4); // [128][D] z tile, f16

  const int tid = threadIdx.x;

  // ---- async-stage codebook f16 (128KB) + e_sq (4KB): global -> LDS,
  //      no VGPR round-trip, tracked by ASYNCcnt. le sits at LDS offset 0.
  {
    const unsigned chunks = (KQ * DQ * 2u) / 16u;   // 8192 x 16B
    for (unsigned i = tid; i < chunks; i += 256u) {
      unsigned off = i * 16u;                        // same LDS + global offset
      asm volatile("global_load_async_to_lds_b128 %0, %1, %2 offset:0"
                   :: "v"(off), "v"(off), "s"(embF16) : "memory");
    }
    unsigned eoff = (unsigned)(KQ * DQ * 2) + (unsigned)tid * 16u;
    unsigned goff = (unsigned)tid * 16u;             // 256 x 16B = 4KB
    asm volatile("global_load_async_to_lds_b128 %0, %1, %2 offset:0"
                 :: "v"(eoff), "v"(goff), "s"(esqg) : "memory");
  }

  // ---- stage 128-row z tile (coalesced over hw), convert to f16 in VALU;
  //      overlaps with the async codebook copy above.
  const int row0 = blockIdx.x * 128;
  const int bimg = row0 >> 12;          // row0 / 4096
  const int hw0  = row0 & (HWQ - 1);    // 128 | 4096, never crosses image
  const float* zbase = zg + (size_t)bimg * (DQ * HWQ) + hw0;
  for (int i = tid; i < 128 * DQ; i += 256) {
    int d = i >> 7, r = i & 127;
    lz[r * DQ + d] = (f16)zbase[(size_t)d * HWQ + r];
  }

  // async copies must land before any thread reads le/esq
  asm volatile("s_wait_asynccnt 0x0" ::: "memory");
  __syncthreads();

  // ---- per-wave WMMA sweep over the codebook ----
  const int wave = tid >> 5, lane = tid & 31;
  const int m = lane & 15, half = lane >> 4;
  const int wrow0 = wave * 16;

  // A fragment (16x32 f16, contiguous k order, same as B): lane holds row M=lane%16;
  // elements are one contiguous run: d = 32c + 16*half + [0..15].
  const f16* zp = lz + (wrow0 + m) * DQ;
  v16h a0, a1;
  a0 = *(const v16h*)(zp + 16 * half);
  a1 = *(const v16h*)(zp + 32 + 16 * half);

  const v8f vzero = {0.f, 0.f, 0.f, 0.f, 0.f, 0.f, 0.f, 0.f};
  float minv[8]; int mini[8];
  #pragma unroll
  for (int r = 0; r < 8; ++r) { minv[r] = 3.4e38f; mini[r] = 0; }

  // Two independent 16-code tiles per iteration: the two WMMA accumulator
  // chains interleave, filling the 4-slot WMMA->VALU hazard window with
  // the other tile's work instead of v_nops.
  for (int kt = 0; kt < KQ / 16; kt += 2) {
    // B fragment (32x16): lane holds column N=lane%16; d = 32c + 16*half + e.
    const f16* epA = le + (kt * 16 + m) * DQ;
    const f16* epB = epA + 16 * DQ;
    v16h b0A = *(const v16h*)(epA + 16 * half);
    v16h b1A = *(const v16h*)(epA + 32 + 16 * half);
    v16h b0B = *(const v16h*)(epB + 16 * half);
    v16h b1B = *(const v16h*)(epB + 32 + 16 * half);

    union { v8f v; float f[8]; } accA, accB;
    accA.v = vzero;
    accB.v = vzero;
    accA.v = __builtin_amdgcn_wmma_f32_16x16x32_f16(false, a0, false, b0A,
                                                    (short)0, accA.v, false, false);
    accB.v = __builtin_amdgcn_wmma_f32_16x16x32_f16(false, a0, false, b0B,
                                                    (short)0, accB.v, false, false);
    accA.v = __builtin_amdgcn_wmma_f32_16x16x32_f16(false, a1, false, b1A,
                                                    (short)0, accA.v, false, false);
    accB.v = __builtin_amdgcn_wmma_f32_16x16x32_f16(false, a1, false, b1B,
                                                    (short)0, accB.v, false, false);

    const int codeA = kt * 16 + m;
    const int codeB = codeA + 16;
    const float esA = esq[codeA];
    const float esB = esq[codeB];
    #pragma unroll
    for (int r = 0; r < 8; ++r) {
      // z_sq is row-constant -> irrelevant to argmin.
      float scoreA = esA - 2.0f * accA.f[r];
      if (scoreA < minv[r]) { minv[r] = scoreA; mini[r] = codeA; }
      float scoreB = esB - 2.0f * accB.f[r];
      if (scoreB < minv[r]) { minv[r] = scoreB; mini[r] = codeB; }
    }
  }

  // ---- cross-lane argmin over the 16 lanes holding the code dimension ----
  #pragma unroll
  for (int r = 0; r < 8; ++r) {
    float v = minv[r]; int ix = mini[r];
    #pragma unroll
    for (int off = 1; off < 16; off <<= 1) {
      float ov = __shfl_xor(v, off, 32);
      int   oi = __shfl_xor(ix, off, 32);
      if (ov < v || (ov == v && oi < ix)) { v = ov; ix = oi; }
    }
    if (m == 0) {
      int row = row0 + wrow0 + r + 8 * half;   // C layout: M = r + 8*half
      idxWs[row] = ix;
      outIdxF[row] = (float)ix;
      atomicAdd(&counts[ix], 1.0f);
    }
  }
}

// ---------------------------------------------------------------------------
// Kernel 2: gather e_q (transposed store), loss partial, embed_sum scatter.
//  block = 256 threads, one row n per thread, d-loop coalesced over n.
// ---------------------------------------------------------------------------
__global__ void k_scatter(const float* __restrict__ zg,
                          const float* __restrict__ emb,
                          const int*   __restrict__ idxWs,
                          float* __restrict__ eqOut,
                          float* __restrict__ esum,
                          float* __restrict__ lossacc)
{
  const int tid = threadIdx.x;
  const int n = blockIdx.x * 256 + tid;
  const int b = n >> 12, hw = n & (HWQ - 1);
  const int code = idxWs[n];
  const float* zrow = zg    + (size_t)b * (DQ * HWQ) + hw;
  float*       erow = eqOut + (size_t)b * (DQ * HWQ) + hw;

  float lsum = 0.0f;
  #pragma unroll 4
  for (int d = 0; d < DQ; ++d) {
    float zv = zrow[(size_t)d * HWQ];
    float ev = emb[code * DQ + d];
    erow[(size_t)d * HWQ] = ev;           // straight-through fwd value == e_q
    float df = ev - zv;
    lsum += df * df;
    atomicAdd(&esum[code * DQ + d], zv);  // one-hot^T @ z
  }

  __shared__ float red[256];
  red[tid] = lsum; __syncthreads();
  for (int s = 128; s > 0; s >>= 1) {
    if (tid < s) red[tid] += red[tid + s];
    __syncthreads();
  }
  if (tid == 0) atomicAdd(lossacc, red[0]);
}

// ---------------------------------------------------------------------------
// Kernel 3: EMA buffers, Laplace smoothing, dead reinit, stats, vq_loss.
//  one block, 1024 threads (one per code); LDS tree reductions for sums.
// ---------------------------------------------------------------------------
__device__ __forceinline__ float block_sum_1024(float v, float* red) {
  int t = threadIdx.x;
  red[t] = v; __syncthreads();
  for (int s = 512; s > 0; s >>= 1) {
    if (t < s) red[t] += red[t + s];
    __syncthreads();
  }
  float r = red[0]; __syncthreads();
  return r;
}

__global__ void k_ema(const float* __restrict__ zg,
                      const float* __restrict__ cluster_size,
                      const float* __restrict__ embed_avg,
                      const float* __restrict__ counts,
                      const float* __restrict__ esum,
                      const float* __restrict__ lossacc,
                      float* __restrict__ outLoss,
                      float* __restrict__ outStats,
                      float* __restrict__ outEmb,
                      float* __restrict__ outCs,
                      float* __restrict__ outEa,
                      int N)
{
  __shared__ float red[1024];
  const int k = threadIdx.x;

  const float cnt = counts[k];
  const float csb = cluster_size[k] * DECAYQ + cnt * (1.0f - DECAYQ);
  outCs[k] = csb;

  const float n = block_sum_1024(csb, red);
  const float sumcs = n + (float)KQ * EPSQ;       // sum(csb + EPS)
  const float cs = (csb + EPSQ) / fmaxf(sumcs, 1e-8f) * n;
  const bool dead = cs < 1.0f;
  const float csf = dead ? 1.0f : cs;

  // deterministic PRNG stand-in for jax.random.randint(key(1), ...)
  unsigned s = (unsigned)k * 747796405u + 2891336453u;
  s ^= s >> 16; s *= 2246822519u; s ^= s >> 13;
  const int ridx = (int)(s % (unsigned)N);
  const int rb = ridx >> 12, rhw = ridx & (HWQ - 1);

  for (int d = 0; d < DQ; ++d) {
    float ea = embed_avg[k * DQ + d] * DECAYQ + esum[k * DQ + d] * (1.0f - DECAYQ);
    float eab = dead ? zg[(size_t)rb * (DQ * HWQ) + (size_t)d * HWQ + rhw] : ea;
    outEa[k * DQ + d]  = eab;
    outEmb[k * DQ + d] = eab / csf;
  }

  const float p = cnt / (float)N;
  const float H = block_sum_1024(p * logf(p + 1e-10f), red);
  const float totuse = block_sum_1024(cnt > 0.0f ? 1.0f : 0.0f, red);

  if (k == 0) {
    outStats[0] = expf(-H);                  // perplexity
    outStats[1] = totuse;                    // cluster_use
    // codebook + commit losses are numerically (1+beta)*mean(diff^2)
    outLoss[0] = lossacc[0] * (1.0f + BETAQ) / ((float)N * (float)DQ);
  }
}

// ---------------------------------------------------------------------------
// Launch. Inputs: z_e[N*D] f32, embedding[K*D] f32, cluster_size[K] f32,
// embed_avg[K*D] f32.  d_out = e_q | indices | vq_loss | stats(2) |
// embedding_new | cs_buf | ea_buf  (flat f32, return order).
// ---------------------------------------------------------------------------
extern "C" void kernel_launch(void* const* d_in, const int* in_sizes, int n_in,
                              void* d_out, int out_size, void* d_ws, size_t ws_size,
                              hipStream_t stream) {
  (void)n_in; (void)out_size; (void)ws_size;
  const float* zg   = (const float*)d_in[0];
  const float* emb  = (const float*)d_in[1];
  const float* csin = (const float*)d_in[2];
  const float* eain = (const float*)d_in[3];

  const int N = in_sizes[0] / DQ;            // 65536

  float* out = (float*)d_out;
  const size_t off_idx   = (size_t)N * DQ;
  const size_t off_loss  = off_idx + N;
  const size_t off_stats = off_loss + 1;
  const size_t off_emb   = off_stats + 2;
  const size_t off_cs    = off_emb + (size_t)KQ * DQ;
  const size_t off_ea    = off_cs + KQ;

  // ws (floats): counts[K] | esum[K*D] | loss[1] | idx[N] | pad | embF16 | esq
  float* ws      = (float*)d_ws;
  float* counts  = ws;
  float* esum    = ws + KQ;
  float* lossacc = ws + KQ + KQ * DQ;
  int*   idxWs   = (int*)(ws + KQ + KQ * DQ + 1);
  size_t f16Off  = ((size_t)KQ + KQ * DQ + 1 + N + 3) & ~(size_t)3; // 16B align
  f16*   embF16  = (f16*)(ws + f16Off);
  float* esqws   = ws + f16Off + (KQ * DQ) / 2;
  const int nzero = KQ + KQ * DQ + 1;

  k_init<<<(nzero + 255) / 256, 256, 0, stream>>>(ws, nzero);
  k_embf16<<<(KQ * DQ) / 256, 256, 0, stream>>>(emb, embF16, esqws);

  const size_t ldsBytes = (size_t)KQ * DQ * 2 + KQ * 4 + 128 * DQ * 2; // 148KB
  k_argmin<<<N / 128, 256, ldsBytes, stream>>>(zg, embF16, esqws,
                                               out + off_idx, idxWs, counts);

  k_scatter<<<N / 256, 256, 0, stream>>>(zg, emb, idxWs, out, esum, lossacc);

  k_ema<<<1, 1024, 0, stream>>>(zg, csin, eain, counts, esum, lossacc,
                                out + off_loss, out + off_stats,
                                out + off_emb, out + off_cs, out + off_ea, N);
}
